// MultiHeadSelfAttention_14035953123532
// MI455X (gfx1250) — matrix operand order repaired and checked by reading, not compiled, not run on hardware
//
#include <hip/hip_runtime.h>
#include <math.h>

// ---------------------------------------------------------------------------
// MHSA forward for MI455X (gfx1250, wave32, WMMA bf16 16x16x32, f32 accum).
// B=2, N=4096, C=512, H=8, Dh=64.
// Workspace requirement: ~44 MB (x_bf16 + w_bf16 + Q/K/V + attn_out, all bf16).
// ---------------------------------------------------------------------------

#define BATCH   2
#define SEQ     4096
#define CH      512
#define NH      8
#define DH      64
#define MTOT    (BATCH * SEQ)      // 8192 rows
#define QKV_N   (3 * CH)           // 1536
#define KSTEPS  (CH / 32)          // 16

typedef unsigned short u16;
typedef __attribute__((ext_vector_type(16))) __bf16        v16bf;
typedef __attribute__((ext_vector_type(8)))  float         v8f;
typedef __attribute__((ext_vector_type(4)))  unsigned int  v4u;

union Frag {
    v16bf bf;
    v4u   q[2];
    u16   h[16];
};

__device__ __forceinline__ u16 f2bf(float f) {
    unsigned u = __builtin_bit_cast(unsigned, f);
    u += 0x7FFFu + ((u >> 16) & 1u);          // round-to-nearest-even
    return (u16)(u >> 16);
}

// A-matrix fragment (16x32 bf16), source row-major with leading dim `ld`.
// Lane holds row (lane&15); element e: K = (e<8 ? lh*8+e : 16+lh*8+(e-8)).
__device__ __forceinline__ v16bf load_fragA(const u16* base, int ld, int lane) {
    const int row = lane & 15;
    const int lh  = (lane >> 4) & 1;
    const u16* p = base + (size_t)row * ld + lh * 8;
    Frag f;
    f.q[0] = *(const v4u*)(p);        // K = lh*8 + 0..7
    f.q[1] = *(const v4u*)(p + 16);   // K = 16 + lh*8 + 0..7
    return f.bf;
}

// B-matrix fragment (32x16 bf16). B[k][n] read from src[n*ld + k] (source
// stores one output-column per row, contiguous in K).
// Lane holds column (lane&15); element e: K = (e<8 ? lh*8+e : 16+lh*8+(e-8)).
__device__ __forceinline__ v16bf load_fragB(const u16* base, int ld, int lane) {
    const int n  = lane & 15;
    const int lh = (lane >> 4) & 1;
    const u16* p = base + (size_t)n * ld + lh * 8;
    Frag f;
    f.q[0] = *(const v4u*)(p);
    f.q[1] = *(const v4u*)(p + 16);
    return f.bf;
}

__device__ __forceinline__ v8f wmma_bf16(v16bf a, v16bf b, v8f c) {
    return __builtin_amdgcn_wmma_f32_16x16x32_bf16(
        /*neg_a=*/false, a, /*neg_b=*/false, b,
        /*c_mod=*/(short)0, c, /*reuse_a=*/false, /*reuse_b=*/false);
}

// CDNA5 async global -> LDS copy (16B per lane), tracked by ASYNCcnt.
__device__ __forceinline__ void async_ld_b128(unsigned lds_off,
                                              const void* gaddr) {
    unsigned long long ga = (unsigned long long)(size_t)gaddr;
    asm volatile("global_load_async_to_lds_b128 %0, %1, off"
                 :: "v"(lds_off), "v"(ga) : "memory");
}
__device__ __forceinline__ void wait_asynccnt0() {
    asm volatile("s_wait_asynccnt 0x0" ::: "memory");
}

// ---------------------------------------------------------------------------
// Kernel 0: fp32 -> bf16 conversion
// ---------------------------------------------------------------------------
__global__ void to_bf16_kernel(const float* __restrict__ src,
                               u16* __restrict__ dst, int n) {
    int i = blockIdx.x * blockDim.x + threadIdx.x;
    if (i < n) dst[i] = f2bf(src[i]);
}

// ---------------------------------------------------------------------------
// Kernels 1/3 GEMM core: wave owns a 32x64 output strip (2 A-frags x 4
// B-frags -> 8 WMMAs per 32-deep k-step).  The k-loop is FULLY UNROLLED with
// a ping-pong fragment buffer (index s&1 is a compile-time constant), so the
// next step's loads are issued before the current step's WMMAs without any
// register-rotation movs.
// Block = 256 threads = 8 waves -> block tile 256 x 64.
// ---------------------------------------------------------------------------
__device__ __forceinline__ void gemm_wave_32x64(
    const u16* __restrict__ abase,   // A tile origin: row mt, k 0 (ld = CH)
    const u16* __restrict__ bbase,   // B tile origin: col nt, k 0 (ld = CH)
    int lane, v8f acc[2][4]) {

    v16bf a[2][2], b[2][4];
#pragma unroll
    for (int i = 0; i < 2; ++i)
        a[0][i] = load_fragA(abase + (size_t)(i * 16) * CH, CH, lane);
#pragma unroll
    for (int j = 0; j < 4; ++j)
        b[0][j] = load_fragB(bbase + (size_t)(j * 16) * CH, CH, lane);

#pragma unroll
    for (int s = 0; s < KSTEPS; ++s) {
        const int cur = s & 1;
        const int nxt = cur ^ 1;
        if (s + 1 < KSTEPS) {               // prefetch step s+1 fragments
            const int kn = (s + 1) * 32;
#pragma unroll
            for (int i = 0; i < 2; ++i)
                a[nxt][i] = load_fragA(abase + (size_t)(i * 16) * CH + kn, CH, lane);
#pragma unroll
            for (int j = 0; j < 4; ++j)
                b[nxt][j] = load_fragB(bbase + (size_t)(j * 16) * CH + kn, CH, lane);
        }
#pragma unroll
        for (int i = 0; i < 2; ++i)
#pragma unroll
            for (int j = 0; j < 4; ++j)
                acc[i][j] = wmma_bf16(a[cur][i], b[cur][j], acc[i][j]);
    }
}

// Kernel 1: QKV projection. qkv = x @ w_qkv^T (M=8192, N=1536, K=512),
// output scattered to head-major Q/K/V: [b][h][n][d], bf16.
__global__ __launch_bounds__(256) void qkv_gemm_kernel(
    const u16* __restrict__ xb,     // [MTOT][CH] bf16
    const u16* __restrict__ wb,     // [QKV_N][CH] bf16
    u16* __restrict__ qw, u16* __restrict__ kw, u16* __restrict__ vw) {

    const int lane = threadIdx.x & 31;
    const int wv   = threadIdx.x >> 5;
    const int mt   = blockIdx.y * 256 + wv * 32;
    const int nt   = blockIdx.x * 64;
    const int lh   = (lane >> 4) & 1;
    const int n15  = lane & 15;

    v8f acc[2][4] = {};
    gemm_wave_32x64(xb + (size_t)mt * CH, wb + (size_t)nt * CH, lane, acc);

#pragma unroll
    for (int i = 0; i < 2; ++i)
#pragma unroll
    for (int j = 0; j < 4; ++j)
#pragma unroll
    for (int r = 0; r < 8; ++r) {
        const int row = mt + i * 16 + r + 8 * lh;
        const int col = nt + j * 16 + n15;            // 0..1535
        const int b   = row >> 12;
        const int n   = row & (SEQ - 1);
        const int s   = col >> 9;                     // 0=Q 1=K 2=V
        const int rem = col & (CH - 1);
        const int h   = rem >> 6;
        const int d   = rem & (DH - 1);
        u16* dst = (s == 0) ? qw : (s == 1) ? kw : vw;
        dst[(((size_t)b * NH + h) * SEQ + n) * DH + d] = f2bf(acc[i][j][r]);
    }
}

// ---------------------------------------------------------------------------
// Kernel 2: flash attention per (b,h).  Block = 8 waves x 16 query rows.
// K tile (32x64) staged via CDNA5 async global->LDS DMA; V^T tile (64x32)
// staged through VGPRs (transpose); online softmax; P re-read via LDS.
// ---------------------------------------------------------------------------
__global__ __launch_bounds__(256) void attn_kernel(
    const u16* __restrict__ qw, const u16* __restrict__ kw,
    const u16* __restrict__ vw, u16* __restrict__ attn) {

    __shared__ u16 kt[32 * DH];        // [key][d]      4 KB
    __shared__ u16 vt[DH * 32];        // [d][key]      4 KB
    __shared__ u16 pl[8 * 16 * 32];    // per-wave P    8 KB

    const int tid  = threadIdx.x;
    const int lane = tid & 31;
    const int wv   = tid >> 5;
    const int lh   = (lane >> 4) & 1;
    const int n15  = lane & 15;
    const int bh   = blockIdx.x;                       // 0..15
    const int q0   = blockIdx.y * 128 + wv * 16;

    const u16* Q = qw + (size_t)bh * SEQ * DH;
    const u16* K = kw + (size_t)bh * SEQ * DH;
    const u16* V = vw + (size_t)bh * SEQ * DH;

    // Q fragments are invariant over the key loop: load once.
    const v16bf aq0 = load_fragA(Q + (size_t)q0 * DH,      DH, lane);
    const v16bf aq1 = load_fragA(Q + (size_t)q0 * DH + 32, DH, lane);

    v8f o[4] = {};
    float m[8], l[8];
#pragma unroll
    for (int r = 0; r < 8; ++r) { m[r] = -INFINITY; l[r] = 0.0f; }

    const unsigned kt_base = (unsigned)(size_t)(&kt[0]);

    for (int kt0 = 0; kt0 < SEQ; kt0 += 32) {
        __syncthreads();                               // previous tile fully read
        // ---- stage K tile (32x64 bf16 = 4 KB) via async DMA: 16 B/thread ----
        async_ld_b128(kt_base + (unsigned)tid * 16u,
                      K + (size_t)kt0 * DH + (size_t)tid * 8);
        // ---- stage V tile transposed through VGPRs: vt[d][key] ----
        {
            const int key = tid >> 3;                  // 0..31
            const int d0  = (tid & 7) * 8;             // 0,8,..,56
            union { v4u q; u16 h[8]; } vu;
            vu.q = *(const v4u*)(V + (size_t)(kt0 + key) * DH + d0);
#pragma unroll
            for (int e = 0; e < 8; ++e) vt[(d0 + e) * 32 + key] = vu.h[e];
        }
        // prefetch next K tile while we compute on this one
        if (kt0 + 32 < SEQ)
            __builtin_prefetch(K + (size_t)(kt0 + 32) * DH + tid * 8, 0, 0);
        wait_asynccnt0();                              // this wave's DMA done
        __syncthreads();                               // all waves' tiles visible

        // ---- S = Q K^T for two 16-key subtiles ----
        v8f s[2];
#pragma unroll
        for (int j = 0; j < 2; ++j) {
            v16bf bk0 = load_fragB(kt + (size_t)(j * 16) * DH,      DH, lane);
            v16bf bk1 = load_fragB(kt + (size_t)(j * 16) * DH + 32, DH, lane);
            v8f sj = {};
            sj = wmma_bf16(aq0, bk0, sj);
            sj = wmma_bf16(aq1, bk1, sj);
            s[j] = sj;
        }

        // ---- online softmax (rows live across 16-lane halves) ----
        const float scale = 0.125f;                    // Dh^-0.5
#pragma unroll
        for (int r = 0; r < 8; ++r) {
            float v0 = s[0][r] * scale;
            float v1 = s[1][r] * scale;
            float mx = fmaxf(v0, v1);
#pragma unroll
            for (int msk = 1; msk < 16; msk <<= 1)
                mx = fmaxf(mx, __shfl_xor(mx, msk, 32));
            const float mnew = fmaxf(m[r], mx);
            const float corr = __expf(m[r] - mnew);
            const float p0 = __expf(v0 - mnew);
            const float p1 = __expf(v1 - mnew);
            float rs = p0 + p1;
#pragma unroll
            for (int msk = 1; msk < 16; msk <<= 1)
                rs += __shfl_xor(rs, msk, 32);
            l[r] = l[r] * corr + rs;
            m[r] = mnew;
#pragma unroll
            for (int t = 0; t < 4; ++t) o[t][r] *= corr;
            // stash P (bf16) in this wave's LDS region, D-layout -> row-major
            const int prow = r + 8 * lh;
            pl[wv * 512 + prow * 32 + n15]      = f2bf(p0);
            pl[wv * 512 + prow * 32 + 16 + n15] = f2bf(p1);
        }
        // same-wave DS ops are in-order (ISA 15.15); stop compiler reordering
        asm volatile("" ::: "memory");

        // ---- O += P @ V ----
        v16bf ap = load_fragA(pl + wv * 512, 32, lane);
#pragma unroll
        for (int t = 0; t < 4; ++t) {
            v16bf bv = load_fragB(vt + (size_t)(t * 16) * 32, 32, lane);
            o[t] = wmma_bf16(ap, bv, o[t]);
        }
    }

    // ---- normalize and emit attn_out in [B,N,C] bf16 ----
    const int b = bh >> 3, h = bh & (NH - 1);
#pragma unroll
    for (int r = 0; r < 8; ++r) {
        const float inv = 1.0f / l[r];
        const int n = q0 + r + 8 * lh;
#pragma unroll
        for (int t = 0; t < 4; ++t) {
            const int c = h * DH + t * 16 + n15;
            attn[((size_t)(b * SEQ + n)) * CH + c] = f2bf(o[t][r] * inv);
        }
    }
}

// ---------------------------------------------------------------------------
// Kernel 3: output projection.  out = attn @ w_proj^T + b  (M=8192,N=512,K=512)
// ---------------------------------------------------------------------------
__global__ __launch_bounds__(256) void proj_gemm_kernel(
    const u16* __restrict__ ab,     // [MTOT][CH] bf16
    const u16* __restrict__ wb,     // [CH][CH] bf16
    const float* __restrict__ bias,
    float* __restrict__ out) {

    const int lane = threadIdx.x & 31;
    const int wv   = threadIdx.x >> 5;
    const int mt   = blockIdx.y * 256 + wv * 32;
    const int nt   = blockIdx.x * 64;
    const int lh   = (lane >> 4) & 1;
    const int n15  = lane & 15;

    v8f acc[2][4] = {};
    gemm_wave_32x64(ab + (size_t)mt * CH, wb + (size_t)nt * CH, lane, acc);

#pragma unroll
    for (int i = 0; i < 2; ++i)
#pragma unroll
    for (int j = 0; j < 4; ++j)
#pragma unroll
    for (int r = 0; r < 8; ++r) {
        const int row = mt + i * 16 + r + 8 * lh;
        const int col = nt + j * 16 + n15;
        out[(size_t)row * CH + col] = acc[i][j][r] + bias[col];
    }
}

// ---------------------------------------------------------------------------
extern "C" void kernel_launch(void* const* d_in, const int* in_sizes, int n_in,
                              void* d_out, int out_size, void* d_ws, size_t ws_size,
                              hipStream_t stream) {
    const float* x      = (const float*)d_in[0];   // [2,4096,512]
    const float* w_qkv  = (const float*)d_in[1];   // [1536,512]
    const float* w_proj = (const float*)d_in[2];   // [512,512]
    const float* b_proj = (const float*)d_in[3];   // [512]
    float* out = (float*)d_out;

    // Workspace carve-up (bf16 elements)
    u16* xb    = (u16*)d_ws;                               // 8192*512
    u16* wqkvb = xb    + (size_t)MTOT * CH;                // 1536*512
    u16* wprojb= wqkvb + (size_t)QKV_N * CH;               // 512*512
    u16* qw    = wprojb+ (size_t)CH * CH;                  // 2*8*4096*64
    u16* kw    = qw    + (size_t)BATCH * NH * SEQ * DH;
    u16* vw    = kw    + (size_t)BATCH * NH * SEQ * DH;
    u16* attn  = vw    + (size_t)BATCH * NH * SEQ * DH;    // 8192*512

    // 0) fp32 -> bf16
    {
        int n0 = MTOT * CH, n1 = QKV_N * CH, n2 = CH * CH;
        to_bf16_kernel<<<(n0 + 255) / 256, 256, 0, stream>>>(x, xb, n0);
        to_bf16_kernel<<<(n1 + 255) / 256, 256, 0, stream>>>(w_qkv, wqkvb, n1);
        to_bf16_kernel<<<(n2 + 255) / 256, 256, 0, stream>>>(w_proj, wprojb, n2);
    }
    // 1) QKV projection:  grid (1536/64, 8192/256)
    qkv_gemm_kernel<<<dim3(QKV_N / 64, MTOT / 256), 256, 0, stream>>>(
        xb, wqkvb, qw, kw, vw);
    // 2) flash attention: grid (B*H, 4096/128)
    attn_kernel<<<dim3(BATCH * NH, SEQ / 128), 256, 0, stream>>>(
        qw, kw, vw, attn);
    // 3) output projection: grid (512/64, 8192/256)
    proj_gemm_kernel<<<dim3(CH / 64, MTOT / 256), 256, 0, stream>>>(
        attn, wprojb, b_proj, out);
}
